// ASCA_4337916969253
// MI455X (gfx1250) — matrix operand order repaired and checked by reading, not compiled, not run on hardware
//
#include <hip/hip_runtime.h>
#include <hip/hip_bf16.h>

// ---------------------------------------------------------------------------
// ASCA block (transposed attention + conv-FFN) for gfx1250.
// GEMM-shaped work (1x1 convs, implicit-GEMM 3x3 convs, attention Grams) runs
// on v_wmma_f32_16x16x32_bf16 with double-buffered LDS staged through
// global_load_async_to_lds_b128 (ASYNCcnt-tracked async copy).
// ---------------------------------------------------------------------------

#define NPIX 16384   // H*W = 128*128
#define WIMG 128
#define HIMG 128
#define CCH  192
#define BB   4
#define HEADS 6

typedef __attribute__((ext_vector_type(16))) __bf16 v16bf;
typedef __attribute__((ext_vector_type(8)))  float  v8f;

union Frag { v16bf v; unsigned int u[8]; };
union Pack2 { ushort2 s; unsigned int u; };

#define ASYNC_LDS_B128(lds, gaddr) \
  asm volatile("global_load_async_to_lds_b128 %0, %1, off" \
               :: "v"(lds), "v"(gaddr) : "memory")
#define WAIT_ASYNC() asm volatile("s_wait_asynccnt 0" ::: "memory")

__device__ __forceinline__ unsigned short f2bf(float f) {
  unsigned int u = __float_as_uint(f);
  u = (u + 0x7FFFu + ((u >> 16) & 1u)) >> 16;   // round-to-nearest-even
  return (unsigned short)u;
}
__device__ __forceinline__ float bf2f(unsigned short h) {
  return __uint_as_float(((unsigned int)h) << 16);
}

// ---------------------------------------------------------------------------
// Weight conversion kernels (f32 -> bf16, with padding/reordering for WMMA)
// ---------------------------------------------------------------------------
__global__ void k_f2bf(const float* __restrict__ s, unsigned short* __restrict__ d, int n) {
  int i = blockIdx.x * blockDim.x + threadIdx.x;
  if (i < n) d[i] = f2bf(s[i]);
}

// sel_in_w [510][192][3][3] -> [9][512][192] (rows 510,511 zero)
__global__ void k_cvt_selin(const float* __restrict__ s, unsigned short* __restrict__ d) {
  int i = blockIdx.x * blockDim.x + threadIdx.x;
  if (i >= 9 * 512 * 192) return;
  int ci = i % 192;
  int co = (i / 192) % 512;
  int t  = i / (192 * 512);
  int ky = t / 3, kx = t % 3;
  d[i] = (co < 510) ? f2bf(s[((co * 192 + ci) * 3 + ky) * 3 + kx]) : (unsigned short)0;
}

// sel_out_w [192][510][3][3] -> [9][192][512] (cols 510,511 zero)
__global__ void k_cvt_selout(const float* __restrict__ s, unsigned short* __restrict__ d) {
  int i = blockIdx.x * blockDim.x + threadIdx.x;
  if (i >= 9 * 192 * 512) return;
  int ck = i % 512;
  int co = (i / 512) % 192;
  int t  = i / (512 * 192);
  int ky = t / 3, kx = t % 3;
  d[i] = (ck < 510) ? f2bf(s[((co * 510 + ck) * 3 + ky) * 3 + kx]) : (unsigned short)0;
}

// ---------------------------------------------------------------------------
// Channels-first LayerNorm over C=192; optionally also emit norm+pos_embed.
// ---------------------------------------------------------------------------
__global__ __launch_bounds__(256)
void k_ln(const float* __restrict__ x, const float* __restrict__ lw,
          const float* __restrict__ lb, unsigned short* __restrict__ out,
          const float* __restrict__ pos, unsigned short* __restrict__ outpos) {
  int idx = blockIdx.x * blockDim.x + threadIdx.x;      // 0..B*N-1
  int b = idx / NPIX, n = idx % NPIX;
  size_t base = (size_t)b * CCH * NPIX + n;
  float s = 0.f, s2 = 0.f;
  for (int c = 0; c < CCH; ++c) {
    float v = x[base + (size_t)c * NPIX];
    s += v; s2 += v * v;
  }
  float mean = s * (1.0f / CCH);
  float var  = s2 * (1.0f / CCH) - mean * mean;
  float rstd = rsqrtf(var + 1e-6f);
  for (int c = 0; c < CCH; ++c) {
    size_t o = base + (size_t)c * NPIX;
    float v  = (x[o] - mean) * rstd * lw[c] + lb[c];
    out[o] = f2bf(v);
    if (outpos) outpos[o] = f2bf(v + pos[c]);
  }
}

// ---------------------------------------------------------------------------
// Generic WMMA GEMM: Out[b] = sum_t A[t] @ shift(B[b], t)   (implicit conv)
//   A: [TAPS][M][KC] bf16, B: [B][KC][NPIX] bf16; TAPS 1 (1x1) or 9 (3x3 SAME)
//   MODE 0: bf16 store; 1: OutF = Res + alpha*acc; 2: OutF = Res + (1-a)*acc
// 64x64 block tile, 8 waves x 2 accumulators, K-step 32, double-buffered LDS
// fed by global_load_async_to_lds_b128 overlapping the WMMAs. All tap/K/mode
// control flow is compile-time (template), so the hot loop has no divisions
// or dynamic branching beyond the uniform dy-row validity test.
// ---------------------------------------------------------------------------
template <int TAPS, int KC, int MODE>
__global__ __launch_bounds__(256)
void k_gemm(const unsigned short* __restrict__ A,
            const unsigned short* __restrict__ Bsrc,
            int M,
            const float* __restrict__ Res, const float* __restrict__ alphaPtr,
            unsigned short* __restrict__ OutB, float* __restrict__ OutF) {
  constexpr int KSTEPS = KC >> 5;
  constexpr int NSTEPS = TAPS * KSTEPS;
  static_assert((NSTEPS & 1) == 0, "pipeline assumes even step count");

  __shared__ alignas(16) unsigned short Als[2][64 * 40];  // [row][k], pad 40
  __shared__ alignas(16) unsigned short Bls[2][32 * 72];  // [k][n],  pad 72

  const int tid  = threadIdx.x;
  const int lane = tid & 31;
  const int wave = tid >> 5;
  const int b    = blockIdx.z;
  const int m0   = blockIdx.y * 64;
  const int n0   = blockIdx.x * 64;

  const int m_sub  = wave >> 1;
  const int n_sub0 = (wave & 1) * 2;
  const int kh     = lane >> 4;

  v8f acc0 = {0.f,0.f,0.f,0.f,0.f,0.f,0.f,0.f};
  v8f acc1 = acc0;

  const int ar  = tid >> 2;          // A loader: row 0..63
  const int akb = (tid & 3) * 8;     // A loader: k base (halves)
  const int bk  = tid >> 3;          // B loader: k 0..31
  const int bjb = (tid & 7) * 8;     // B loader: n base within tile

  const size_t bBatch = (size_t)b * KC * NPIX;

  auto stage = [&](int s, int bufi) {
    const int t  = (TAPS == 1) ? 0 : s / KSTEPS;          // const-div
    const int k0 = (TAPS == 1) ? (s << 5) : ((s % KSTEPS) << 5);
    // ---- A tile (contiguous, 16B/thread) : async copy to LDS ----
    {
      const unsigned short* src = A + ((size_t)t * M + (m0 + ar)) * KC + (k0 + akb);
      unsigned lds = (unsigned)(size_t)&Als[bufi][ar * 40 + akb];
      ASYNC_LDS_B128(lds, (unsigned long long)(size_t)src);
    }
    // ---- B tile ----
    const int dy = (TAPS == 1) ? 0 : (t / 3 - 1);
    const int dx = (TAPS == 1) ? 0 : (t % 3 - 1);
    const int sy = (n0 >> 7) + dy;                  // uniform over the tile
    unsigned ldsB = (unsigned)(size_t)&Bls[bufi][bk * 72 + bjb];
    if (TAPS == 1 || (unsigned)sy < (unsigned)HIMG) {
      const unsigned short* rp =
          Bsrc + bBatch + (size_t)(k0 + bk) * NPIX + ((size_t)sy << 7);
      const int xb = (n0 & (WIMG - 1)) + bjb;
      if (TAPS == 1 || dx == 0) {
        ASYNC_LDS_B128(ldsB, (unsigned long long)(size_t)(rp + xb));
      } else {
        unsigned short vv[8];
        #pragma unroll
        for (int j = 0; j < 8; ++j) {               // branch-free clip
          const int sx  = xb + j + dx;
          const int sxc = sx < 0 ? 0 : (sx > WIMG - 1 ? WIMG - 1 : sx);
          const unsigned short v = rp[sxc];
          vv[j] = ((unsigned)sx < (unsigned)WIMG) ? v : (unsigned short)0;
        }
        *(uint4*)&Bls[bufi][bk * 72 + bjb] = *(const uint4*)vv;
      }
    } else {
      uint4 z; z.x = z.y = z.z = z.w = 0u;
      *(uint4*)&Bls[bufi][bk * 72 + bjb] = z;
    }
  };

  auto compute = [&](int bufi) {
    Frag af, bf0, bf1;
    const int arow = m_sub * 16 + (lane & 15);
    #pragma unroll
    for (int p = 0; p < 8; ++p) {                   // 2x ds_load_b128
      const int kk = ((p >> 2) * 16) + kh * 8 + (p & 3) * 2;
      af.u[p] = *(const unsigned int*)&Als[bufi][arow * 40 + kk];
    }
    const int col = n_sub0 * 16 + (lane & 15);
    #pragma unroll
    for (int p = 0; p < 8; ++p) {                   // d16 lo/hi pair loads
      const int k = ((p >> 2) * 16) + kh * 8 + (p & 3) * 2;
      Pack2 t0, t1;
      t0.s.x = Bls[bufi][k * 72 + col];
      t0.s.y = Bls[bufi][(k + 1) * 72 + col];
      t1.s.x = Bls[bufi][k * 72 + col + 16];
      t1.s.y = Bls[bufi][(k + 1) * 72 + col + 16];
      bf0.u[p] = t0.u;
      bf1.u[p] = t1.u;
    }
    acc0 = __builtin_amdgcn_wmma_f32_16x16x32_bf16(false, af.v, false, bf0.v,
                                                   (short)0, acc0, false, false);
    acc1 = __builtin_amdgcn_wmma_f32_16x16x32_bf16(false, af.v, false, bf1.v,
                                                   (short)0, acc1, false, false);
  };

  // ---- software pipeline (x2 unrolled so buffer index is compile-time) ----
  stage(0, 0);
  WAIT_ASYNC();
  __syncthreads();
  for (int s = 0; s < NSTEPS; s += 2) {
    stage(s + 1, 1);
    compute(0);
    WAIT_ASYNC();
    __syncthreads();
    if (s + 2 < NSTEPS) stage(s + 2, 0);
    compute(1);
    WAIT_ASYNC();
    __syncthreads();
  }

  // ---- epilogue ----
  const int mrow  = m0 + m_sub * 16 + ((lane >= 16) ? 8 : 0);
  const int ncol0 = n0 + n_sub0 * 16 + (lane & 15);
  float scale = 0.f;
  if (MODE == 1) scale = alphaPtr[0];
  if (MODE == 2) scale = 1.0f - alphaPtr[0];
  #pragma unroll
  for (int r = 0; r < 8; ++r) {
    const size_t i0 = ((size_t)b * M + (mrow + r)) * NPIX + ncol0;
    const size_t i1 = i0 + 16;
    if (MODE == 0) {
      OutB[i0] = f2bf(acc0[r]);
      OutB[i1] = f2bf(acc1[r]);
    } else {
      OutF[i0] = Res[i0] + scale * acc0[r];
      OutF[i1] = Res[i1] + scale * acc1[r];
    }
  }
}

// ---------------------------------------------------------------------------
// Depthwise 3x3 SAME (bf16 in/out), branch-free clamp+select taps.
// ---------------------------------------------------------------------------
__global__ void k_dw3(const unsigned short* __restrict__ in,
                      const float* __restrict__ w,
                      unsigned short* __restrict__ out, int Cx, long long total) {
  long long idx = (long long)blockIdx.x * blockDim.x + threadIdx.x;
  if (idx >= total) return;
  int n = (int)(idx % NPIX);
  long long cb = idx / NPIX;          // b*Cx + c
  int c = (int)(cb % Cx);
  const unsigned short* src = in + cb * (long long)NPIX;
  int y = n >> 7, x = n & 127;
  float acc = 0.f;
  #pragma unroll
  for (int ky = 0; ky < 3; ++ky) {
    #pragma unroll
    for (int kx = 0; kx < 3; ++kx) {
      int sy = y + ky - 1, sx = x + kx - 1;
      bool ok = ((unsigned)sy < (unsigned)HIMG) & ((unsigned)sx < (unsigned)WIMG);
      int syc = sy < 0 ? 0 : (sy > HIMG - 1 ? HIMG - 1 : sy);
      int sxc = sx < 0 ? 0 : (sx > WIMG - 1 ? WIMG - 1 : sx);
      float v = bf2f(src[(syc << 7) + sxc]);
      acc += ok ? w[c * 9 + ky * 3 + kx] * v : 0.f;
    }
  }
  out[cb * (long long)NPIX + n] = f2bf(acc);
}

// Depthwise 3x3 + SiLU for the FFN (510 real channels padded to 512)
__global__ void k_dw3_silu(const unsigned short* __restrict__ in,
                           const float* __restrict__ w,
                           unsigned short* __restrict__ out) {
  long long idx = (long long)blockIdx.x * blockDim.x + threadIdx.x;
  int n = (int)(idx % NPIX);
  long long cb = idx / NPIX;          // b*512 + c
  int c = (int)(cb % 512);
  if (c >= 510) { out[cb * (long long)NPIX + n] = 0; return; }
  const unsigned short* src = in + cb * (long long)NPIX;
  int y = n >> 7, x = n & 127;
  float acc = 0.f;
  #pragma unroll
  for (int ky = 0; ky < 3; ++ky) {
    #pragma unroll
    for (int kx = 0; kx < 3; ++kx) {
      int sy = y + ky - 1, sx = x + kx - 1;
      bool ok = ((unsigned)sy < (unsigned)HIMG) & ((unsigned)sx < (unsigned)WIMG);
      int syc = sy < 0 ? 0 : (sy > HIMG - 1 ? HIMG - 1 : sy);
      int sxc = sx < 0 ? 0 : (sx > WIMG - 1 ? WIMG - 1 : sx);
      float v = bf2f(src[(syc << 7) + sxc]);
      acc += ok ? w[c * 9 + ky * 3 + kx] * v : 0.f;
    }
  }
  float s = acc / (1.0f + expf(-acc));   // SiLU
  out[cb * (long long)NPIX + n] = f2bf(s);
}

// ---------------------------------------------------------------------------
// Attention Grams via WMMA: G1 = q @ k^T, G2 = pos_q @ pos_k^T  (32x32 per b,h)
// ---------------------------------------------------------------------------
__global__ __launch_bounds__(256)
void k_gram(const unsigned short* __restrict__ q2, const unsigned short* __restrict__ kv2,
            const unsigned short* __restrict__ pq, const unsigned short* __restrict__ pk,
            float* __restrict__ G1, float* __restrict__ G2) {
  const int blk = blockIdx.x;                 // b*HEADS + h
  const int b = blk / HEADS, h = blk % HEADS;
  const int lane = threadIdx.x & 31;
  const int wave = threadIdx.x >> 5;
  const int grp  = wave >> 2;                 // 0: qk, 1: pos
  const int tile = wave & 3;
  const int tm = tile >> 1, tn = tile & 1;

  const unsigned short* Ap;
  const unsigned short* Bp;
  if (grp == 0) {
    Ap = q2  + ((size_t)b * CCH + h * 32) * NPIX;
    Bp = kv2 + ((size_t)b * 2 * CCH + h * 32) * NPIX;     // k = kv channels 0..191
  } else {
    Ap = pq + ((size_t)b * CCH + h * 32) * NPIX;
    Bp = pk + ((size_t)b * CCH + h * 32) * NPIX;
  }
  const int kh = lane >> 4;
  const unsigned short* ap = Ap + (size_t)(tm * 16 + (lane & 15)) * NPIX + kh * 8;
  const unsigned short* bp = Bp + (size_t)(tn * 16 + (lane & 15)) * NPIX + kh * 8;

  v8f acc = {0.f,0.f,0.f,0.f,0.f,0.f,0.f,0.f};
  #pragma unroll 2
  for (int n = 0; n < NPIX; n += 32) {
    Frag af, bf;
    uint4 a0 = *(const uint4*)(ap + n);
    uint4 a1 = *(const uint4*)(ap + n + 16);
    uint4 b0 = *(const uint4*)(bp + n);
    uint4 b1 = *(const uint4*)(bp + n + 16);
    __builtin_prefetch(ap + n + 512, 0, 1);
    af.u[0] = a0.x; af.u[1] = a0.y; af.u[2] = a0.z; af.u[3] = a0.w;
    af.u[4] = a1.x; af.u[5] = a1.y; af.u[6] = a1.z; af.u[7] = a1.w;
    bf.u[0] = b0.x; bf.u[1] = b0.y; bf.u[2] = b0.z; bf.u[3] = b0.w;
    bf.u[4] = b1.x; bf.u[5] = b1.y; bf.u[6] = b1.z; bf.u[7] = b1.w;
    acc = __builtin_amdgcn_wmma_f32_16x16x32_bf16(false, af.v, false, bf.v,
                                                  (short)0, acc, false, false);
  }
  float* G = grp ? G2 : G1;
  const int dcol = tn * 16 + (lane & 15);
  const int crow = tm * 16 + ((lane >= 16) ? 8 : 0);
  #pragma unroll
  for (int r = 0; r < 8; ++r)
    G[(size_t)blk * 1024 + (crow + r) * 32 + dcol] = acc[r];
}

// ---------------------------------------------------------------------------
// Finalize: l2-norm folding + temperature + dual softmax; S = attn + pos_attn
// ---------------------------------------------------------------------------
__global__ __launch_bounds__(256)
void k_attn_fin(const unsigned short* __restrict__ q2, const unsigned short* __restrict__ kv2,
                const float* __restrict__ G1, const float* __restrict__ G2,
                const float* __restrict__ temp, float* __restrict__ S) {
  __shared__ float part[256];
  __shared__ float nrm[64];
  __shared__ float sm[2][32][32];
  const int blk = blockIdx.x;
  const int b = blk / HEADS, h = blk % HEADS;
  const int tid = threadIdx.x;

  // phase 1: row l2 norms of q (rows 0..31) and k (rows 32..63), vectorized
  const int rr = tid >> 2, pp = tid & 3;
  const unsigned short* row = (rr < 32)
      ? q2  + ((size_t)b * CCH + h * 32 + rr) * NPIX
      : kv2 + ((size_t)b * 2 * CCH + h * 32 + (rr - 32)) * NPIX;
  const uint4* rp = (const uint4*)(row + pp * 4096);
  float s = 0.f;
  for (int i = 0; i < 512; ++i) {
    uint4 d = rp[i];
    unsigned u[4] = {d.x, d.y, d.z, d.w};
    #pragma unroll
    for (int q = 0; q < 4; ++q) {
      float lo = __uint_as_float(u[q] << 16);
      float hi = __uint_as_float(u[q] & 0xffff0000u);
      s += lo * lo + hi * hi;
    }
  }
  part[tid] = s;
  __syncthreads();
  if (pp == 0) {
    float t = part[tid] + part[tid + 1] + part[tid + 2] + part[tid + 3];
    nrm[rr] = fmaxf(sqrtf(t), 1e-12f);
  }
  __syncthreads();

  // phase 2: softmax rows (which=0: attn w/ temperature & norms; 1: pos_attn)
  if (tid < 64) {
    const int which = tid >> 5, c = tid & 31;
    const float tv = temp[h];
    float rowv[32], mx = -1e30f;
    for (int d = 0; d < 32; ++d) {
      float g = which ? G2[(size_t)blk * 1024 + c * 32 + d]
                      : G1[(size_t)blk * 1024 + c * 32 + d] * tv / (nrm[c] * nrm[32 + d]);
      rowv[d] = g;
      mx = fmaxf(mx, g);
    }
    float sum = 0.f;
    for (int d = 0; d < 32; ++d) { rowv[d] = expf(rowv[d] - mx); sum += rowv[d]; }
    float inv = 1.0f / sum;
    for (int d = 0; d < 32; ++d) sm[which][c][d] = rowv[d] * inv;
  }
  __syncthreads();
  if (tid < 32)
    for (int d = 0; d < 32; ++d)
      S[(size_t)blk * 1024 + tid * 32 + d] = sm[0][tid][d] + sm[1][tid][d];
}

// ---------------------------------------------------------------------------
// out[b, h*32+c, n] = sum_d S[b,h,c,d] * v[b,h,d,n]   (v = kv2 channels 192..)
// ---------------------------------------------------------------------------
__global__ __launch_bounds__(256)
void k_apply(const float* __restrict__ S, const unsigned short* __restrict__ kv2,
             unsigned short* __restrict__ ao) {
  __shared__ float Sr[32];
  const int b = blockIdx.z, c = blockIdx.y;
  const int h = c >> 5, cl = c & 31;
  const int tid = threadIdx.x;
  if (tid < 32) Sr[tid] = S[((size_t)(b * HEADS + h) * 32 + cl) * 32 + tid];
  __syncthreads();
  const int n = blockIdx.x * 256 + tid;
  const unsigned short* vbase = kv2 + ((size_t)b * 2 * CCH + CCH + h * 32) * NPIX + n;
  float acc = 0.f;
  #pragma unroll 8
  for (int d = 0; d < 32; ++d) acc += Sr[d] * bf2f(vbase[(size_t)d * NPIX]);
  ao[((size_t)b * CCH + c) * NPIX + n] = f2bf(acc);
}

// ---------------------------------------------------------------------------
extern "C" void kernel_launch(void* const* d_in, const int* in_sizes, int n_in,
                              void* d_out, int out_size, void* d_ws, size_t ws_size,
                              hipStream_t stream) {
  (void)in_sizes; (void)n_in; (void)out_size; (void)ws_size;
  const float* x        = (const float*)d_in[0];
  const float* ln_w     = (const float*)d_in[1];
  const float* ln_b     = (const float*)d_in[2];
  const float* temp     = (const float*)d_in[3];
  const float* q_w      = (const float*)d_in[4];
  const float* q_dw_w   = (const float*)d_in[5];
  const float* kv_w     = (const float*)d_in[6];
  const float* kv_dw_w  = (const float*)d_in[7];
  const float* proj_w   = (const float*)d_in[8];
  const float* pos_e    = (const float*)d_in[9];
  const float* pos_q_w  = (const float*)d_in[10];
  const float* pos_k_w  = (const float*)d_in[11];
  const float* sel_in_w = (const float*)d_in[12];
  const float* sel_dw_w = (const float*)d_in[13];
  const float* sel_o_w  = (const float*)d_in[14];
  const float* alpha    = (const float*)d_in[15];

  const size_t BN = (size_t)BB * CCH * NPIX;     // 12,582,912
  const size_t BG = (size_t)BB * 512 * NPIX;     // 33,554,432

  char* cur = (char*)d_ws;
  auto alloc = [&](size_t bytes) -> void* {
    void* p = cur; cur += (bytes + 255) & ~(size_t)255; return p;
  };
  unsigned short* normx = (unsigned short*)alloc(BN * 2);     // reused as nx later
  unsigned short* posx  = (unsigned short*)alloc(BN * 2);
  unsigned short* q1    = (unsigned short*)alloc(BN * 2);
  unsigned short* q2    = (unsigned short*)alloc(BN * 2);
  unsigned short* kv1   = (unsigned short*)alloc(2 * BN * 2);
  unsigned short* kv2   = (unsigned short*)alloc(2 * BN * 2);
  unsigned short* pq    = (unsigned short*)alloc(BN * 2);
  unsigned short* pk    = (unsigned short*)alloc(BN * 2);
  unsigned short* ao    = (unsigned short*)alloc(BN * 2);
  float*          x1    = (float*)alloc(BN * 4);
  unsigned short* g     = (unsigned short*)alloc(BG * 2);
  unsigned short* g2    = (unsigned short*)alloc(BG * 2);
  float* G1 = (float*)alloc(BB * HEADS * 1024 * 4);
  float* G2 = (float*)alloc(BB * HEADS * 1024 * 4);
  float* Smat = (float*)alloc(BB * HEADS * 1024 * 4);
  unsigned short* wq    = (unsigned short*)alloc(192 * 192 * 2);
  unsigned short* wkv   = (unsigned short*)alloc(384 * 192 * 2);
  unsigned short* wpq   = (unsigned short*)alloc(192 * 192 * 2);
  unsigned short* wpk   = (unsigned short*)alloc(192 * 192 * 2);
  unsigned short* wproj = (unsigned short*)alloc(192 * 192 * 2);
  unsigned short* wsi   = (unsigned short*)alloc((size_t)9 * 512 * 192 * 2);
  unsigned short* wso   = (unsigned short*)alloc((size_t)9 * 192 * 512 * 2);

  // --- weight conversion ---
  k_f2bf<<<(192*192 + 255) / 256, 256, 0, stream>>>(q_w, wq, 192 * 192);
  k_f2bf<<<(384*192 + 255) / 256, 256, 0, stream>>>(kv_w, wkv, 384 * 192);
  k_f2bf<<<(192*192 + 255) / 256, 256, 0, stream>>>(pos_q_w, wpq, 192 * 192);
  k_f2bf<<<(192*192 + 255) / 256, 256, 0, stream>>>(pos_k_w, wpk, 192 * 192);
  k_f2bf<<<(192*192 + 255) / 256, 256, 0, stream>>>(proj_w, wproj, 192 * 192);
  k_cvt_selin <<<(9*512*192) / 256, 256, 0, stream>>>(sel_in_w, wsi);
  k_cvt_selout<<<(9*192*512) / 256, 256, 0, stream>>>(sel_o_w, wso);

  // --- LayerNorm (+pos_embed variant) ---
  k_ln<<<(BB * NPIX) / 256, 256, 0, stream>>>(x, ln_w, ln_b, normx, pos_e, posx);

  // --- 1x1 conv GEMMs (WMMA) ---
  k_gemm<1,192,0><<<dim3(NPIX/64, 3, BB), 256, 0, stream>>>(wq,  normx, 192, nullptr, nullptr, q1,  nullptr);
  k_gemm<1,192,0><<<dim3(NPIX/64, 6, BB), 256, 0, stream>>>(wkv, normx, 384, nullptr, nullptr, kv1, nullptr);
  k_gemm<1,192,0><<<dim3(NPIX/64, 3, BB), 256, 0, stream>>>(wpq, posx,  192, nullptr, nullptr, pq,  nullptr);
  k_gemm<1,192,0><<<dim3(NPIX/64, 3, BB), 256, 0, stream>>>(wpk, posx,  192, nullptr, nullptr, pk,  nullptr);

  // --- depthwise 3x3 on q, kv ---
  k_dw3<<<(unsigned)(BN / 256), 256, 0, stream>>>(q1,  q_dw_w,  q2,  192, (long long)BN);
  k_dw3<<<(unsigned)(2 * BN / 256), 256, 0, stream>>>(kv1, kv_dw_w, kv2, 384, (long long)(2 * BN));

  // --- attention: Grams (WMMA) -> softmax combine -> apply ---
  k_gram<<<BB * HEADS, 256, 0, stream>>>(q2, kv2, pq, pk, G1, G2);
  k_attn_fin<<<BB * HEADS, 256, 0, stream>>>(q2, kv2, G1, G2, temp, Smat);
  k_apply<<<dim3(NPIX / 256, CCH, BB), 256, 0, stream>>>(Smat, kv2, ao);

  // --- proj (1x1 GEMM) fused with residual: x1 = x + alpha*proj(out) ---
  k_gemm<1,192,1><<<dim3(NPIX/64, 3, BB), 256, 0, stream>>>(wproj, ao, 192, x, alpha, nullptr, x1);

  // --- second LayerNorm (reuse normx buffer as nx) ---
  k_ln<<<(BB * NPIX) / 256, 256, 0, stream>>>(x1, ln_w, ln_b, normx, nullptr, nullptr);

  // --- sel_in: 3x3 conv 192->512(pad) as 9-tap implicit GEMM (WMMA) ---
  k_gemm<9,192,0><<<dim3(NPIX/64, 8, BB), 256, 0, stream>>>(wsi, normx, 512, nullptr, nullptr, g, nullptr);

  // --- depthwise 3x3 + SiLU on 510 channels (padded to 512) ---
  k_dw3_silu<<<(unsigned)(BG / 256), 256, 0, stream>>>(g, sel_dw_w, g2);

  // --- sel_out: 3x3 conv 512->192 implicit GEMM + residual (1-alpha) -> d_out
  k_gemm<9,512,2><<<dim3(NPIX/64, 3, BB), 256, 0, stream>>>(wso, g2, 192, x1, alpha, nullptr, (float*)d_out);
}
